// TransformerDecoderBlock_v2_10608569221192
// MI455X (gfx1250) — matrix operand order repaired and checked by reading, not compiled, not run on hardware
//
#include <hip/hip_runtime.h>
#include <hip/hip_bf16.h>
#include <math.h>

// ---------------------------------------------------------------------------
// MI455X (gfx1250) implementation, round 3.
//  - All GEMMs via v_wmma_f32_16x16x32_f16 (f32 accumulate).
//  - B (weights) pre-swizzled at f32->f16 conversion time into WMMA fragment
//    order: [n_tile][k_tile][lane][16 f16] -> each lane loads its whole B
//    fragment as one contiguous 32B global load (L2-resident, coalesced);
//    no LDS / transpose / barrier on the B path.
//  - A staged in LDS row-major (b128 both directions), BK=64 so each barrier
//    pair covers 8 WMMAs.  Unconditional speculative global_prefetch of the
//    next A tile (branch-free staging).
//  Layouts used (A/B: contiguous k per lane half, not the ISA table order):
//    A 16x32 f16: lane<16 -> K 0..15, lane>=16 -> K 16..31
//    B 32x16 f16: lane<16 -> col=lane, K kblk+0..15 (kblk = (lane>=16)*16)
//    C/D 16x16 f32: VGPR r, lane<16 -> M=r, lane>=16 -> M=r+8, N=lane&15
// ---------------------------------------------------------------------------

typedef __attribute__((ext_vector_type(16))) _Float16 v16h;
typedef __attribute__((ext_vector_type(8)))  _Float16 v8h;
typedef __attribute__((ext_vector_type(8)))  float    v8f;

#define DIMC   384
#define NHEADS 12
#define HDIM   32
#define NTOK   6144      // 16*16*24
#define BATCH  2
#define MROWS  (BATCH * NTOK)   // 12288

// ---------------- A fragment loader (from LDS, row-major) ----------------
__device__ __forceinline__ v16h load_a_frag(const _Float16* base, int ld, int lane) {
    const _Float16* p = base + (size_t)(lane & 15) * ld + ((lane >> 4) << 4);
    v8h lo = *(const v8h*)(p);        // K = half*16 + 0..7
    v8h hi = *(const v8h*)(p + 8);    // K = half*16 + 8..15
    return __builtin_shufflevector(lo, hi, 0,1,2,3,4,5,6,7,8,9,10,11,12,13,14,15);
}

__device__ __forceinline__ v8f wmma_f16(v16h a, v16h b, v8f c) {
    return __builtin_amdgcn_wmma_f32_16x16x32_f16(false, a, false, b, (short)0, c,
                                                  false, false);
}

// ---------------- GEMM: C = A(f16,MxK) * Bswz(f16) + bias, epilogue -------
// EPI 0: outF[idx] = val                      (f32, e.g. qkv buffer)
// EPI 1: outF[idx] = resid[idx] + val         (residual update of h)
// EPI 2: outH[idx] = f16(gelu_exact(val))     (fc1 -> act buffer)
template <int EPI>
__global__ __launch_bounds__(256, 2)
void gemm_wmma_kernel(const _Float16* __restrict__ A, const _Float16* __restrict__ Bp,
                      const float* __restrict__ bias, const float* __restrict__ resid,
                      float* __restrict__ outF, _Float16* __restrict__ outH,
                      int N, int K) {
    constexpr int BM = 128, BK = 64, LDP = BK + 8;   // padded A row: 72 f16 = 144B
    __shared__ alignas(16) _Float16 As[BM][LDP];

    const int t     = threadIdx.x;
    const int lane  = t & 31;
    const int wave  = t >> 5;
    const int wm    = wave & 3;       // 4 strips of 32 rows
    const int wn    = wave >> 2;      // 2 strips of 32 cols (2 n-tiles each)
    const int bm0   = blockIdx.y * BM;
    const int bn0   = blockIdx.x * 64;
    const int ktile = K >> 5;                    // K/32 tiles
    const int tn0   = (blockIdx.x << 2) + (wn << 1);  // first 16-col tile of strip

    v8f acc[2][2] = {};

    for (int kk = 0; kk < K; kk += BK) {
        // ---- stage A tile: 128x64 f16 = 1024 8-elt chunks, 4 per thread ----
        #pragma unroll
        for (int i = 0; i < 4; ++i) {
            int chunk = t + i * 256;
            int r = chunk >> 3;
            int c = (chunk & 7) * 8;
            const _Float16* gp = A + (size_t)(bm0 + r) * K + kk + c;
            v8h v = *(const v8h*)gp;
            *(v8h*)(&As[r][c]) = v;
            // speculative prefetch of next K tile; harmless past the end
            __builtin_prefetch(gp + BK, 0, 2);
        }
        __syncthreads();

        // ---- two 32-K WMMA steps per stage ----
        #pragma unroll
        for (int ks = 0; ks < 2; ++ks) {
            int tk = (kk >> 5) + ks;
            // B fragments: one contiguous 32B load per lane, coalesced
            v16h b0 = *(const v16h*)(Bp + ((((size_t)tn0 + 0) * ktile + tk) * 32 + lane) * 16);
            v16h b1 = *(const v16h*)(Bp + ((((size_t)tn0 + 1) * ktile + tk) * 32 + lane) * 16);
            v16h a0 = load_a_frag(&As[wm * 32 +  0][ks * 32], LDP, lane);
            v16h a1 = load_a_frag(&As[wm * 32 + 16][ks * 32], LDP, lane);

            acc[0][0] = wmma_f16(a0, b0, acc[0][0]);
            acc[0][1] = wmma_f16(a0, b1, acc[0][1]);
            acc[1][0] = wmma_f16(a1, b0, acc[1][0]);
            acc[1][1] = wmma_f16(a1, b1, acc[1][1]);
        }
        __syncthreads();
    }

    // ---- epilogue ----
    #pragma unroll
    for (int mi = 0; mi < 2; ++mi) {
        #pragma unroll
        for (int ni = 0; ni < 2; ++ni) {
            int gcol  = bn0 + wn * 32 + ni * 16 + (lane & 15);
            int rbase = bm0 + wm * 32 + mi * 16 + ((lane >> 4) << 3);
            float bv  = bias[gcol];
            #pragma unroll
            for (int r = 0; r < 8; ++r) {
                size_t idx = (size_t)(rbase + r) * N + gcol;
                float val  = acc[mi][ni][r] + bv;
                if (EPI == 0) {
                    outF[idx] = val;
                } else if (EPI == 1) {
                    outF[idx] = resid[idx] + val;
                } else { // exact GELU -> f16
                    float g = 0.5f * val * (1.0f + erff(val * 0.70710678118654752f));
                    outH[idx] = (_Float16)g;
                }
            }
        }
    }
}

// ---------------- weight f32 -> f16 with WMMA-fragment swizzle -------------
// dst index = ((tn*ktile + tk)*32 + lane)*16 + e
//   col n = tn*16 + (lane&15);  k = tk*32 + ((lane>>4)<<4) + e
__global__ void pack_w16_kernel(const float* __restrict__ src, _Float16* __restrict__ dst,
                                int N, int Ktiles, int total) {
    int i = blockIdx.x * 256 + threadIdx.x;
    if (i >= total) return;
    int e    = i & 15;
    int lane = (i >> 4) & 31;
    int tmp  = i >> 9;
    int tk   = tmp % Ktiles;
    int tn   = tmp / Ktiles;
    int n = tn * 16 + (lane & 15);
    int k = tk * 32 + ((lane >> 4) << 4) + e;
    dst[i] = (_Float16)src[(size_t)k * N + n];
}

// ---------------- LayerNorm (f32 row -> f16 row), one wave32 per row -------
__global__ __launch_bounds__(256)
void ln_f16_kernel(const float* __restrict__ x, const float* __restrict__ g,
                   const float* __restrict__ b, _Float16* __restrict__ y) {
    int lane = threadIdx.x & 31;
    int row  = blockIdx.x * 8 + (threadIdx.x >> 5);
    const float* xr = x + (size_t)row * DIMC;

    float v[12];
    float s = 0.0f;
    #pragma unroll
    for (int i = 0; i < 12; ++i) { v[i] = xr[lane + i * 32]; s += v[i]; }
    #pragma unroll
    for (int o = 16; o > 0; o >>= 1) s += __shfl_xor(s, o, 32);
    float mean = s * (1.0f / DIMC);

    float var = 0.0f;
    #pragma unroll
    for (int i = 0; i < 12; ++i) { float d = v[i] - mean; var += d * d; }
    #pragma unroll
    for (int o = 16; o > 0; o >>= 1) var += __shfl_xor(var, o, 32);
    float rstd = rsqrtf(var * (1.0f / DIMC) + 1e-5f);

    _Float16* yr = y + (size_t)row * DIMC;
    #pragma unroll
    for (int i = 0; i < 12; ++i) {
        int c = lane + i * 32;
        yr[c] = (_Float16)((v[i] - mean) * rstd * g[c] + b[c]);
    }
}

// ---------------- per-token linear attention core, one wave32 per token ----
// qkv rows: [q(h,d) | k(h,d) | v(h,d)], each 384.  softmax(k) over heads (h).
// ctx[d][e] = sum_h k[h][d] v[h][e];  out[h][e] = scale * sum_d q[h][d] ctx[d][e]
// Output written per reference's transpose(0,2,1,3).reshape scramble:
// flat-in-batch offset f = h*(NTOK*32) + n*32 + e.
__global__ __launch_bounds__(256)
void attn_kernel(const float* __restrict__ qkv, _Float16* __restrict__ outH) {
    __shared__ float sQ[8][DIMC];
    __shared__ float sK[8][DIMC];
    __shared__ float sV[8][DIMC];

    int lane = threadIdx.x & 31;
    int w    = threadIdx.x >> 5;
    int tkn  = blockIdx.x * 8 + w;

    const float* p = qkv + (size_t)tkn * (3 * DIMC);
    #pragma unroll
    for (int i = 0; i < 12; ++i) {
        int c = lane + i * 32;
        sQ[w][c] = p[c];
        sK[w][c] = p[DIMC + c];
        sV[w][c] = p[2 * DIMC + c];
    }
    __syncthreads();

    // softmax over heads for dim d = lane
    {
        float kv[NHEADS];
        float mx = -3.4e38f;
        #pragma unroll
        for (int h = 0; h < NHEADS; ++h) {
            kv[h] = sK[w][h * 32 + lane];
            mx = fmaxf(mx, kv[h]);
        }
        float s = 0.0f;
        #pragma unroll
        for (int h = 0; h < NHEADS; ++h) { kv[h] = __expf(kv[h] - mx); s += kv[h]; }
        float inv = 1.0f / s;
        #pragma unroll
        for (int h = 0; h < NHEADS; ++h) sK[w][h * 32 + lane] = kv[h] * inv;
    }
    __syncthreads();

    float vv[NHEADS];
    #pragma unroll
    for (int h = 0; h < NHEADS; ++h) vv[h] = sV[w][h * 32 + lane];

    float ctx[HDIM];  // ctx[d] for e = lane
    #pragma unroll
    for (int d = 0; d < HDIM; ++d) {
        float c = 0.0f;
        #pragma unroll
        for (int h = 0; h < NHEADS; ++h) c += sK[w][h * 32 + d] * vv[h];
        ctx[d] = c;
    }

    int b = tkn / NTOK;
    int n = tkn - b * NTOK;
    size_t obase = (size_t)b * NTOK * DIMC + (size_t)n * 32 + lane;
    const float scale = 0.17677669529663688f;  // HEAD_DIM^-0.5
    #pragma unroll
    for (int h = 0; h < NHEADS; ++h) {
        float o = 0.0f;
        #pragma unroll
        for (int d = 0; d < HDIM; ++d) o += sQ[w][h * 32 + d] * ctx[d];
        outH[obase + (size_t)h * (NTOK * 32)] = (_Float16)(o * scale);
    }
}

// ---------------- transposes (B,C,N) <-> (B,N,C), 32x32 LDS tiles ----------
__global__ void transpose_in_kernel(const float* __restrict__ x, float* __restrict__ h) {
    __shared__ float tile[32][33];
    int b  = blockIdx.z;
    int n0 = blockIdx.x * 32;
    int c0 = blockIdx.y * 32;
    const float* xb = x + (size_t)b * DIMC * NTOK;
    float*       hb = h + (size_t)b * DIMC * NTOK;
    int tx = threadIdx.x, ty = threadIdx.y;
    #pragma unroll
    for (int j = 0; j < 32; j += 8)
        tile[ty + j][tx] = xb[(size_t)(c0 + ty + j) * NTOK + n0 + tx];
    __syncthreads();
    #pragma unroll
    for (int j = 0; j < 32; j += 8)
        hb[(size_t)(n0 + ty + j) * DIMC + c0 + tx] = tile[tx][ty + j];
}

__global__ void transpose_out_kernel(const float* __restrict__ h, float* __restrict__ y) {
    __shared__ float tile[32][33];
    int b  = blockIdx.z;
    int n0 = blockIdx.x * 32;
    int c0 = blockIdx.y * 32;
    const float* hb = h + (size_t)b * DIMC * NTOK;
    float*       yb = y + (size_t)b * DIMC * NTOK;
    int tx = threadIdx.x, ty = threadIdx.y;
    #pragma unroll
    for (int j = 0; j < 32; j += 8)
        tile[ty + j][tx] = hb[(size_t)(n0 + ty + j) * DIMC + c0 + tx];
    __syncthreads();
    #pragma unroll
    for (int j = 0; j < 32; j += 8)
        yb[(size_t)(c0 + ty + j) * NTOK + n0 + tx] = tile[tx][ty + j];
}

// ---------------------------------------------------------------------------
extern "C" void kernel_launch(void* const* d_in, const int* in_sizes, int n_in,
                              void* d_out, int out_size, void* d_ws, size_t ws_size,
                              hipStream_t stream) {
    const float* x        = (const float*)d_in[0];
    // d_in[1] = enc_feats (unused by the reference)
    const float* n1_g     = (const float*)d_in[2];
    const float* n1_b     = (const float*)d_in[3];
    const float* sa_qkv_w = (const float*)d_in[4];
    const float* sa_qkv_b = (const float*)d_in[5];
    const float* sa_prj_w = (const float*)d_in[6];
    const float* sa_prj_b = (const float*)d_in[7];
    const float* n2_g     = (const float*)d_in[8];
    const float* n2_b     = (const float*)d_in[9];
    const float* ca_qkv_w = (const float*)d_in[10];
    const float* ca_qkv_b = (const float*)d_in[11];
    const float* ca_prj_w = (const float*)d_in[12];
    const float* ca_prj_b = (const float*)d_in[13];
    const float* n3_g     = (const float*)d_in[14];
    const float* n3_b     = (const float*)d_in[15];
    const float* fc1_w    = (const float*)d_in[16];
    const float* fc1_b    = (const float*)d_in[17];
    const float* fc2_w    = (const float*)d_in[18];
    const float* fc2_b    = (const float*)d_in[19];
    float* out = (float*)d_out;

    // workspace layout
    char* ws = (char*)d_ws;
    size_t off = 0;
    auto take = [&](size_t bytes) {
        char* p = ws + off;
        off = (off + bytes + 255) & ~(size_t)255;
        return p;
    };
    float*    h      = (float*)   take((size_t)MROWS * DIMC * 4);   // residual stream
    _Float16* a16    = (_Float16*)take((size_t)MROWS * DIMC * 2);   // LN output
    float*    qkvbuf = (float*)   take((size_t)MROWS * 3 * DIMC * 4);
    _Float16* att16  = (_Float16*)take((size_t)MROWS * DIMC * 2);
    _Float16* act16  = (_Float16*)take((size_t)MROWS * 4 * DIMC * 2);
    _Float16* wq16   = (_Float16*)take((size_t)DIMC * 3 * DIMC * 2);
    _Float16* wp16   = (_Float16*)take((size_t)DIMC * DIMC * 2);
    _Float16* wf1    = (_Float16*)take((size_t)DIMC * 4 * DIMC * 2);
    _Float16* wf2    = (_Float16*)take((size_t)4 * DIMC * DIMC * 2);
    (void)ws_size; (void)n_in; (void)in_sizes; (void)out_size;

    dim3 tb(32, 8);
    dim3 tg(NTOK / 32, DIMC / 32, BATCH);
    dim3 gemm_blk(256);
    dim3 g_qkv(3 * DIMC / 64, MROWS / 128);   // (18, 96)
    dim3 g_prj(DIMC / 64,     MROWS / 128);   // (6, 96)
    dim3 g_fc1(4 * DIMC / 64, MROWS / 128);   // (24, 96)
    int rows8 = MROWS / 8;                    // 1536 blocks for LN / attn

    const int n_qkv = 3 * DIMC * DIMC;        // 442368
    const int n_prj = DIMC * DIMC;            // 147456
    const int n_fc  = 4 * DIMC * DIMC;        // 589824
    const int kt384  = DIMC / 32;             // 12
    const int kt1536 = 4 * DIMC / 32;         // 48

    // (B,C,N) -> (B,N,C)
    transpose_in_kernel<<<tg, tb, 0, stream>>>(x, h);

    // ---------------- self-attention block ----------------
    pack_w16_kernel<<<(n_qkv + 255) / 256, 256, 0, stream>>>(sa_qkv_w, wq16, 3 * DIMC, kt384, n_qkv);
    pack_w16_kernel<<<(n_prj + 255) / 256, 256, 0, stream>>>(sa_prj_w, wp16, DIMC, kt384, n_prj);
    ln_f16_kernel<<<rows8, 256, 0, stream>>>(h, n1_g, n1_b, a16);
    gemm_wmma_kernel<0><<<g_qkv, gemm_blk, 0, stream>>>(a16, wq16, sa_qkv_b, nullptr,
                                                        qkvbuf, nullptr, 3 * DIMC, DIMC);
    attn_kernel<<<rows8, 256, 0, stream>>>(qkvbuf, att16);
    gemm_wmma_kernel<1><<<g_prj, gemm_blk, 0, stream>>>(att16, wp16, sa_prj_b, h,
                                                        h, nullptr, DIMC, DIMC);

    // ---------------- "cross"-attention block (same stream input) ----------
    pack_w16_kernel<<<(n_qkv + 255) / 256, 256, 0, stream>>>(ca_qkv_w, wq16, 3 * DIMC, kt384, n_qkv);
    pack_w16_kernel<<<(n_prj + 255) / 256, 256, 0, stream>>>(ca_prj_w, wp16, DIMC, kt384, n_prj);
    ln_f16_kernel<<<rows8, 256, 0, stream>>>(h, n2_g, n2_b, a16);
    gemm_wmma_kernel<0><<<g_qkv, gemm_blk, 0, stream>>>(a16, wq16, ca_qkv_b, nullptr,
                                                        qkvbuf, nullptr, 3 * DIMC, DIMC);
    attn_kernel<<<rows8, 256, 0, stream>>>(qkvbuf, att16);
    gemm_wmma_kernel<1><<<g_prj, gemm_blk, 0, stream>>>(att16, wp16, ca_prj_b, h,
                                                        h, nullptr, DIMC, DIMC);

    // ---------------- MLP block ----------------
    pack_w16_kernel<<<(n_fc + 255) / 256, 256, 0, stream>>>(fc1_w, wf1, 4 * DIMC, kt384, n_fc);
    pack_w16_kernel<<<(n_fc + 255) / 256, 256, 0, stream>>>(fc2_w, wf2, DIMC, kt1536, n_fc);
    ln_f16_kernel<<<rows8, 256, 0, stream>>>(h, n3_g, n3_b, a16);
    gemm_wmma_kernel<2><<<g_fc1, gemm_blk, 0, stream>>>(a16, wf1, fc1_b, nullptr,
                                                        nullptr, act16, 4 * DIMC, DIMC);
    gemm_wmma_kernel<1><<<g_prj, gemm_blk, 0, stream>>>(act16, wf2, fc2_b, h,
                                                        h, nullptr, DIMC, 4 * DIMC);

    // (B,N,C) -> (B,C,N)
    transpose_out_kernel<<<tg, tb, 0, stream>>>(h, out);
}
